// YINEncoder_30537217475366
// MI455X (gfx1250) — matrix operand order repaired and checked by reading, not compiled, not run on hardware
//
#include <hip/hip_runtime.h>
#include <hip/hip_bf16.h>

typedef __attribute__((ext_vector_type(16))) _Float16 v16h;
typedef __attribute__((ext_vector_type(8)))  float    v8f;
typedef __attribute__((ext_vector_type(8)))  _Float16 half8;

#define WIN      2048
#define HOP      256
#define NFRAMES  682
#define LSIG     176400
#define NTAU     1024
#define PADH     256          // zero pad in front of frame (A needs f[k-240])
#define GLEN     2368         // PADH + WIN + 64 tail zeros
#define NTHREADS 128
#define OUT_DIM  50

union AU { half8 h[2]; v16h v; };
union BU { _Float16 e[16]; v16h v; };

__global__ __launch_bounds__(NTHREADS)
void yin_wmma_kernel(const float* __restrict__ x, float* __restrict__ out)
{
    __shared__ _Float16 gpad[GLEN];       // zero-extended frame, f16
    __shared__ float    csumL[WIN + 1];   // prefix sum of squares (fp32)
    __shared__ float    acfL[NTAU];
    __shared__ float    dfL[NTAU];
    __shared__ float    dsumL[NTAU];
    __shared__ float    part[NTHREADS];

    const int tid = threadIdx.x;
    const int s   = blockIdx.x;   // frame index
    const int b   = blockIdx.y;   // batch index

    // ---- zero halos (and body; frame overwrites body after barrier) ----
    for (int i = tid; i < GLEN; i += NTHREADS) gpad[i] = (_Float16)0.0f;
    __syncthreads();

    const float* fbase = x + (size_t)b * LSIG + (size_t)s * HOP;
    if (s + 1 < NFRAMES) __builtin_prefetch(fbase + WIN, 0, 1);  // global_prefetch_b8

    // ---- cooperative load: 16 floats/thread; stage f16 + local energy ----
    float v[16];
    {
        const float4* fp = (const float4*)fbase;
        float acc = 0.f;
        #pragma unroll
        for (int m = 0; m < 4; ++m) {
            float4 q = fp[tid * 4 + m];
            v[4*m+0] = q.x; v[4*m+1] = q.y; v[4*m+2] = q.z; v[4*m+3] = q.w;
        }
        #pragma unroll
        for (int m = 0; m < 16; ++m) {
            gpad[PADH + 16*tid + m] = (_Float16)v[m];
            acc += v[m] * v[m];
        }
        part[tid] = acc;
    }
    __syncthreads();

    // ---- exclusive scan of 128 per-thread energy partials (wave 0) ----
    if (tid < 32) {
        float p0 = part[4*tid+0], p1 = part[4*tid+1];
        float p2 = part[4*tid+2], p3 = part[4*tid+3];
        float tot = p0 + p1 + p2 + p3;
        float run = tot;
        #pragma unroll
        for (int d = 1; d < 32; d <<= 1) {
            float n = __shfl_up(run, d, 32);
            if (tid >= d) run += n;
        }
        float excl = run - tot;
        part[4*tid+0] = excl;
        part[4*tid+1] = excl + p0;
        part[4*tid+2] = excl + p0 + p1;
        part[4*tid+3] = excl + p0 + p1 + p2;
    }
    __syncthreads();

    // ---- emit csum[0..2048] ----
    {
        float run = part[tid];
        #pragma unroll
        for (int m = 0; m < 16; ++m) { csumL[16*tid + m] = run; run += v[m]*v[m]; }
        if (tid == NTHREADS - 1) csumL[WIN] = run;
    }
    __syncthreads();

    // ---- ACF via WMMA: wave w owns taus [256w, 256w+256) ----
    {
        const int wave    = tid >> 5;
        const int lane    = tid & 31;
        const int taubase = wave * 256;
        // wave-uniform trip bound -> scalar loop, EXEC stays all-ones
        const int kend    = __builtin_amdgcn_readfirstlane(WIN - taubase);

        // A (16x32 f16): lane row M, contiguous K in {Kb..Kb+15}
        const int Arow  = lane & 15;
        const int AKb   = (lane < 16) ? 0 : 16;
        const int abase = PADH + AKb - 16 * Arow;   // multiple of 8 -> b128 aligned
        // B (32x16 f16): lane col N=lane%16, contiguous K run; start folds to +lane
        const int bbase = PADH + taubase + lane;

        auto loadA = [&](AU& A, int k) {
            A.h[0] = *(const half8*)(gpad + abase + k);        // ds_load_b128
            A.h[1] = *(const half8*)(gpad + abase + k + 8);    // ds_load_b128
        };
        auto loadB = [&](BU& Bv, int k) {
            const _Float16* bp = gpad + bbase + k;             // compiler merges to b128
            #pragma unroll
            for (int e = 0; e < 16; ++e) Bv.e[e] = bp[e];
        };

        v8f acc = {};
        AU A0, A1; BU B0, B1;
        // skewed ping-pong pipeline: each buffer reloaded one stage ahead of use,
        // so every WMMA only waits for dscnt<=4 (its own 4 loads), never 0.
        loadA(A0, 0);  loadB(B0, 0);
        loadA(A1, 32); loadB(B1, 32);
        int k = 0;
        for (; k + 64 < kend; k += 64) {     // last iter: k = kend-128
            acc = __builtin_amdgcn_wmma_f32_16x16x32_f16(
                      false, A0.v, false, B0.v, (short)0, acc, false, false);
            loadA(A0, k + 64); loadB(B0, k + 64);
            acc = __builtin_amdgcn_wmma_f32_16x16x32_f16(
                      false, A1.v, false, B1.v, (short)0, acc, false, false);
            loadA(A1, k + 96); loadB(B1, k + 96);   // max read: kend-32 (in-bounds)
        }
        // drain: operands for k = kend-64 and kend-32 already resident
        acc = __builtin_amdgcn_wmma_f32_16x16x32_f16(
                  false, A0.v, false, B0.v, (short)0, acc, false, false);
        acc = __builtin_amdgcn_wmma_f32_16x16x32_f16(
                  false, A1.v, false, B1.v, (short)0, acc, false, false);

        // D: VGPR r, lane L -> (M = r + 8*(L>=16), N = L%16); tau = taubase+16M+N
        const int nn  = lane & 15;
        const int mhi = (lane >> 4) << 3;
        #pragma unroll
        for (int r = 0; r < 8; ++r)
            acfL[taubase + 16 * (r + mhi) + nn] = acc[r];
    }
    __syncthreads();

    // ---- df[tau] = csum[W-tau] + csum[W] - csum[tau] - 2*acf[tau] ----
    {
        const float eW = csumL[WIN];
        #pragma unroll
        for (int r = 0; r < NTAU / NTHREADS; ++r) {
            const int tau = tid + r * NTHREADS;
            dfL[tau] = csumL[WIN - tau] + eW - csumL[tau] - 2.0f * acfL[tau];
        }
    }
    __syncthreads();

    // ---- inclusive scan of df[1..] (df[0] excluded), wave 0 ----
    if (tid < 32) {
        const int base = tid * 32;
        float tot = 0.f;
        for (int m = 0; m < 32; ++m) {
            int idx = base + m;
            tot += (idx == 0) ? 0.f : dfL[idx];
        }
        float run = tot;
        #pragma unroll
        for (int d = 1; d < 32; d <<= 1) {
            float n = __shfl_up(run, d, 32);
            if (tid >= d) run += n;
        }
        float excl = run - tot;
        float r2 = excl;
        for (int m = 0; m < 32; ++m) {
            int idx = base + m;
            r2 += (idx == 0) ? 0.f : dfL[idx];
            dsumL[idx] = r2;
        }
    }
    __syncthreads();

    // ---- cmndf at interpolated lags, 50 outputs ----
    if (tid < OUT_DIM) {
        const double midi = (double)(20 + tid);           // slice [15:65] of midis 5..84
        const double lag  = 22050.0 / (440.0 * exp2((midi - 69.0) / 12.0));
        const double fl = floor(lag), cl = ceil(lag);
        const int fi = (int)fl, ci = (int)cl;             // max 850 < NTAU
        const float frac = (float)(lag - fl);
        const float den  = fmaxf((float)(cl - fl), 1.0f);
        const float clo  = dfL[fi] * (float)fi / (dsumL[fi] + 1e-8f);
        const float chi  = dfL[ci] * (float)ci / (dsumL[ci] + 1e-8f);
        out[((size_t)b * OUT_DIM + tid) * NFRAMES + s] =
            (chi - clo) / den * frac + clo;
    }
}

extern "C" void kernel_launch(void* const* d_in, const int* in_sizes, int n_in,
                              void* d_out, int out_size, void* d_ws, size_t ws_size,
                              hipStream_t stream)
{
    const float* x  = (const float*)d_in[0];
    float* out      = (float*)d_out;
    dim3 grid(NFRAMES, 16);   // one block per (frame, batch)
    yin_wmma_kernel<<<grid, NTHREADS, 0, stream>>>(x, out);
}
